// ABMIL_76235669504305
// MI455X (gfx1250) — matrix operand order repaired and checked by reading, not compiled, not run on hardware
//
#include <hip/hip_runtime.h>
#include <hip/hip_fp16.h>
#include <math.h>

typedef __attribute__((ext_vector_type(16))) _Float16 v16h;
typedef __attribute__((ext_vector_type(8)))  _Float16 v8h;
typedef __attribute__((ext_vector_type(8)))  float    v8f;

// problem constants
static constexpr int cB  = 16;
static constexpr int cN  = 4096;
static constexpr int cD  = 768;
static constexpr int cH  = 12;
static constexpr int cL  = 2;
static constexpr int cK  = 16;
static constexpr int cDH = 64;
static constexpr int cS  = 17;            // K+1
static constexpr int cM  = cB * cS;       // 272 rows, exactly 17 tiles of 16

// ---------------------------------------------------------------- helpers
__device__ inline v16h cat16(v8h a, v8h b) {
  v16h r;
#pragma unroll
  for (int i = 0; i < 8; ++i) { r[i] = a[i]; r[8 + i] = b[i]; }
  return r;
}

__device__ inline v16h ldB16(const _Float16* p) {
  v8h b0 = *(const v8h*)p;
  v8h b1 = *(const v8h*)(p + 16);
  return cat16(b0, b1);
}

__device__ inline v8f wmma_f16(v16h a, v16h b, v8f c) {
  return __builtin_amdgcn_wmma_f32_16x16x32_f16(false, a, false, b, (short)0, c, false, false);
}

__device__ inline float gelu_exact(float x) {
  return 0.5f * x * (1.0f + erff(x * 0.70710678118654752f));
}

// ---------------------------------------------------------------- prep: fold gate LN affine into gW1
// gW1h[n,k] = f16(gW1[n,k]*gln_g[k]); colsum[n]=sum_k gW1[n,k]*gln_g[k];
// biasp[n] = gb1[n] + sum_k gW1[n,k]*gln_b[k]
__global__ __launch_bounds__(256) void prep_gate(
    const float* __restrict__ gW1, const float* __restrict__ gln_g,
    const float* __restrict__ gln_b, const float* __restrict__ gb1,
    _Float16* __restrict__ gW1h, float* __restrict__ colsum, float* __restrict__ biasp) {
  const int n = blockIdx.x, t = threadIdx.x;
  float sg = 0.f, sb = 0.f;
  for (int k = t; k < cD; k += 256) {
    float wv = gW1[n * cD + k];
    float wg = wv * gln_g[k];
    gW1h[n * cD + k] = (_Float16)wg;
    sg += wg;
    sb += wv * gln_b[k];
  }
#pragma unroll
  for (int m = 16; m; m >>= 1) { sg += __shfl_xor(sg, m); sb += __shfl_xor(sb, m); }
  __shared__ float ag[8], ab[8];
  const int wid = t >> 5, ln = t & 31;
  if (ln == 0) { ag[wid] = sg; ab[wid] = sb; }
  __syncthreads();
  if (t == 0) {
    float SG = 0.f, SB = 0.f;
    for (int i = 0; i < 8; ++i) { SG += ag[i]; SB += ab[i]; }
    colsum[n] = SG;
    biasp[n] = gb1[n] + SB;
  }
}

__global__ void f32_to_f16(const float* __restrict__ s, _Float16* __restrict__ d, int n) {
  int i = blockIdx.x * blockDim.x + threadIdx.x;
  if (i < n) d[i] = (_Float16)s[i];
}

// ---------------------------------------------------------------- fused LN + gate GEMM + GELU + sigmoid
// each wave handles 16 instances: stream raw x once, f16 into LDS, WMMA 16x64 over K=768,
// apply (mean,rstd) + folded bias in epilogue.
__global__ __launch_bounds__(64) void gate_kernel(
    const float* __restrict__ x, const _Float16* __restrict__ gW1h,
    const float* __restrict__ colsum, const float* __restrict__ biasp,
    const float* __restrict__ gW2, const float* __restrict__ gb2,
    float* __restrict__ w) {
  constexpr int LDA = 776;                       // halves; 16B-aligned row stride
  __shared__ _Float16 ah[2 * 16 * LDA];          // ~48.5 KB
  __shared__ float    stats[2 * 16 * 2];
  const int wv   = threadIdx.x >> 5;
  const int lane = threadIdx.x & 31;
  const int tile = blockIdx.x * 2 + wv;          // 4096 tiles total
  const int r0   = tile * 16;                    // flat instance row base in [B*N)
  _Float16* A  = ah + wv * 16 * LDA;
  float*    st = stats + wv * 32;

  // stage raw x as f16 + per-row mean/var (single pass over x)
  for (int r = 0; r < 16; ++r) {
    const float4* xr = (const float4*)(x + (size_t)(r0 + r) * cD) + lane * 6;
    float s = 0.f, ss = 0.f;
#pragma unroll
    for (int j = 0; j < 3; ++j) {
      float4 v = xr[2 * j];
      float4 u = xr[2 * j + 1];
      s  += v.x + v.y + v.z + v.w + u.x + u.y + u.z + u.w;
      ss += v.x * v.x + v.y * v.y + v.z * v.z + v.w * v.w
          + u.x * u.x + u.y * u.y + u.z * u.z + u.w * u.w;
      v8h hh;
      hh[0] = (_Float16)v.x; hh[1] = (_Float16)v.y; hh[2] = (_Float16)v.z; hh[3] = (_Float16)v.w;
      hh[4] = (_Float16)u.x; hh[5] = (_Float16)u.y; hh[6] = (_Float16)u.z; hh[7] = (_Float16)u.w;
      *(v8h*)(A + r * LDA + lane * 24 + j * 8) = hh;
    }
#pragma unroll
    for (int m = 16; m; m >>= 1) { s += __shfl_xor(s, m); ss += __shfl_xor(ss, m); }
    if (lane == 0) {
      float mean = s * (1.0f / 768.0f);
      float var  = ss * (1.0f / 768.0f) - mean * mean;
      st[r * 2]     = mean;
      st[r * 2 + 1] = rsqrtf(var + 1e-5f);
    }
  }

  // WMMA: [16 x 768] x [768 x 64]
  const int mr = lane & 15;
  const int hi = lane >> 4;
  v8f c0 = {}, c1 = {}, c2 = {}, c3 = {};
  for (int kc = 0; kc < cD; kc += 32) {
    v8h lo = *(const v8h*)(A + mr * LDA + kc + hi * 8);
    v8h hp = *(const v8h*)(A + mr * LDA + kc + hi * 8 + 16);
    v16h af = cat16(lo, hp);
    const _Float16* wb = gW1h + (size_t)mr * cD + kc + hi * 8;
    c0 = wmma_f16(af, ldB16(wb + (size_t)(0 * 16) * cD), c0);
    c1 = wmma_f16(af, ldB16(wb + (size_t)(1 * 16) * cD), c1);
    c2 = wmma_f16(af, ldB16(wb + (size_t)(2 * 16) * cD), c2);
    c3 = wmma_f16(af, ldB16(wb + (size_t)(3 * 16) * cD), c3);
  }

  // epilogue: apply LN stats, GELU, dot with gW2, sigmoid
  float meanj[8], rstdj[8];
#pragma unroll
  for (int j = 0; j < 8; ++j) {
    int m = j + hi * 8;
    meanj[j] = st[m * 2];
    rstdj[j] = st[m * 2 + 1];
  }
  float part[8];
#pragma unroll
  for (int j = 0; j < 8; ++j) part[j] = 0.f;
#pragma unroll
  for (int nt = 0; nt < 4; ++nt) {
    v8f cc = (nt == 0) ? c0 : (nt == 1) ? c1 : (nt == 2) ? c2 : c3;
    int n = nt * 16 + mr;
    float cs = colsum[n], bp = biasp[n], g2 = gW2[n];
#pragma unroll
    for (int j = 0; j < 8; ++j) {
      float pre = rstdj[j] * cc[j] - meanj[j] * rstdj[j] * cs + bp;
      part[j] += gelu_exact(pre) * g2;
    }
  }
  float b2 = gb2[0];
#pragma unroll
  for (int j = 0; j < 8; ++j) {
#pragma unroll
    for (int msk = 1; msk < 16; msk <<= 1) part[j] += __shfl_xor(part[j], msk);
    if (mr == 0) {
      int m = j + hi * 8;
      w[r0 + m] = 1.0f / (1.0f + expf(-(part[j] + b2)));
    }
  }
}

// ---------------------------------------------------------------- top-k (iterative argmax, stable ties)
__global__ __launch_bounds__(256) void topk_kernel(
    const float* __restrict__ w, float* __restrict__ topv, int* __restrict__ topi) {
  __shared__ float sw[cN];
  __shared__ float rv[256];
  __shared__ int   ri[256];
  const int b = blockIdx.x, t = threadIdx.x;
  for (int i = t; i < cN; i += 256) sw[i] = w[b * cN + i];
  __syncthreads();
  for (int sel = 0; sel < cK; ++sel) {
    float bv = -1e30f; int bi = cN;
    for (int i = t; i < cN; i += 256) {
      float v = sw[i];
      if (v > bv || (v == bv && i < bi)) { bv = v; bi = i; }
    }
    rv[t] = bv; ri[t] = bi;
    __syncthreads();
    for (int s = 128; s > 0; s >>= 1) {
      if (t < s) {
        if (rv[t + s] > rv[t] || (rv[t + s] == rv[t] && ri[t + s] < ri[t])) {
          rv[t] = rv[t + s]; ri[t] = ri[t + s];
        }
      }
      __syncthreads();
    }
    if (t == 0) {
      topv[b * cK + sel] = rv[0];
      topi[b * cK + sel] = ri[0];
      sw[ri[0]] = -1e30f;
    }
    __syncthreads();
  }
}

// ---------------------------------------------------------------- gather selected rows + cls into seq [272 x 768]
__global__ __launch_bounds__(256) void gather_seq(
    const float* __restrict__ x, const float* __restrict__ cls,
    const int* __restrict__ topi, const float* __restrict__ topv, float* __restrict__ seq) {
  const int row = blockIdx.x;
  const int b = row / cS, s = row % cS, t = threadIdx.x;
  const float* src;
  float scale = 1.0f;
  if (s == 0) {
    src = cls;
  } else {
    int idx = topi[b * cK + s - 1];
    scale = topv[b * cK + s - 1];
    src = x + ((size_t)b * cN + idx) * cD;
  }
  for (int j = t; j < cD; j += 256) seq[(size_t)row * cD + j] = src[j] * scale;
}

// ---------------------------------------------------------------- layernorm over rows (optional residual), writes f32+f16
__global__ __launch_bounds__(256) void ln_rows(
    const float* __restrict__ in, const float* __restrict__ resid,
    const float* __restrict__ g, const float* __restrict__ bvec,
    float* __restrict__ out32, _Float16* __restrict__ out16) {
  const int row = blockIdx.x, t = threadIdx.x;
  float vals[3];
  float s = 0.f, ss = 0.f;
#pragma unroll
  for (int j = 0; j < 3; ++j) {
    int idx = t + j * 256;
    float v = in[(size_t)row * cD + idx];
    if (resid) v += resid[(size_t)row * cD + idx];
    vals[j] = v; s += v; ss += v * v;
  }
#pragma unroll
  for (int m = 16; m; m >>= 1) { s += __shfl_xor(s, m); ss += __shfl_xor(ss, m); }
  __shared__ float as_[8], ab_[8], mv[2];
  const int wid = t >> 5, ln = t & 31;
  if (ln == 0) { as_[wid] = s; ab_[wid] = ss; }
  __syncthreads();
  if (t == 0) {
    float S = 0.f, SS = 0.f;
    for (int i = 0; i < 8; ++i) { S += as_[i]; SS += ab_[i]; }
    float mean = S * (1.0f / 768.0f);
    float var  = SS * (1.0f / 768.0f) - mean * mean;
    mv[0] = mean; mv[1] = rsqrtf(var + 1e-5f);
  }
  __syncthreads();
  float mean = mv[0], rstd = mv[1];
#pragma unroll
  for (int j = 0; j < 3; ++j) {
    int idx = t + j * 256;
    float zn = (vals[j] - mean) * rstd * g[idx] + bvec[idx];
    if (out32) out32[(size_t)row * cD + idx] = zn;
    out16[(size_t)row * cD + idx] = (_Float16)zn;
  }
}

// ---------------------------------------------------------------- WMMA GEMM: out[MxN] = A[MxK](f16) * W[NxK]^T(f16) + bias
// one wave per 16x64 output tile
__global__ __launch_bounds__(32) void gemm_f16(
    const _Float16* __restrict__ A, const _Float16* __restrict__ W,
    const float* __restrict__ bias, float* __restrict__ out,
    int Nn, int K, int Mtiles) {
  const int mtile = blockIdx.x % Mtiles;
  const int ng    = blockIdx.x / Mtiles;
  const int lane  = threadIdx.x & 31;
  const int mr = lane & 15, hi = lane >> 4;
  v8f c0 = {}, c1 = {}, c2 = {}, c3 = {};
  const _Float16* arow = A + (size_t)(mtile * 16 + mr) * K;
  for (int kc = 0; kc < K; kc += 32) {
    v16h af = cat16(*(const v8h*)(arow + kc + hi * 8),
                    *(const v8h*)(arow + kc + hi * 8 + 16));
    const _Float16* wb = W + (size_t)(ng * 64 + mr) * K + kc + hi * 8;
    c0 = wmma_f16(af, ldB16(wb + (size_t)(0 * 16) * K), c0);
    c1 = wmma_f16(af, ldB16(wb + (size_t)(1 * 16) * K), c1);
    c2 = wmma_f16(af, ldB16(wb + (size_t)(2 * 16) * K), c2);
    c3 = wmma_f16(af, ldB16(wb + (size_t)(3 * 16) * K), c3);
  }
#pragma unroll
  for (int nt = 0; nt < 4; ++nt) {
    v8f cc = (nt == 0) ? c0 : (nt == 1) ? c1 : (nt == 2) ? c2 : c3;
    int n = ng * 64 + nt * 16 + mr;
    float bb = bias[n];
#pragma unroll
    for (int j = 0; j < 8; ++j) {
      int m = mtile * 16 + j + hi * 8;
      out[(size_t)m * Nn + n] = cc[j] + bb;
    }
  }
}

// ---------------------------------------------------------------- per (bag, head) attention on S=17
__global__ __launch_bounds__(64) void attention_kernel(
    const float* __restrict__ qkv, _Float16* __restrict__ oh) {
  __shared__ float qs[cS * cDH], ks[cS * cDH], vs[cS * cDH], att[cS * 18];
  const int bh = blockIdx.x;
  const int b = bh / cH, h = bh % cH;
  const int t = threadIdx.x;
  for (int s = 0; s < cS; ++s) {
    const float* row = qkv + (size_t)(b * cS + s) * (3 * cD) + h * cDH;
    qs[s * cDH + t] = row[t];
    ks[s * cDH + t] = row[cD + t];
    vs[s * cDH + t] = row[2 * cD + t];
  }
  __syncthreads();
  for (int idx = t; idx < cS * cS; idx += 64) {
    int qi = idx / cS, ki = idx % cS;
    float acc = 0.f;
#pragma unroll 8
    for (int d = 0; d < cDH; ++d) acc += qs[qi * cDH + d] * ks[ki * cDH + d];
    att[qi * 18 + ki] = acc * 0.125f;   // 1/sqrt(64)
  }
  __syncthreads();
  if (t < cS) {
    float mx = -1e30f;
    for (int k = 0; k < cS; ++k) mx = fmaxf(mx, att[t * 18 + k]);
    float sm = 0.f;
    for (int k = 0; k < cS; ++k) { float e = expf(att[t * 18 + k] - mx); att[t * 18 + k] = e; sm += e; }
    float inv = 1.0f / sm;
    for (int k = 0; k < cS; ++k) att[t * 18 + k] *= inv;
  }
  __syncthreads();
  for (int qi = 0; qi < cS; ++qi) {
    float acc = 0.f;
#pragma unroll
    for (int k = 0; k < cS; ++k) acc += att[qi * 18 + k] * vs[k * cDH + t];
    oh[(size_t)(b * cS + qi) * cD + h * cDH + t] = (_Float16)acc;
  }
}

// ---------------------------------------------------------------- final output: cls rows
__global__ __launch_bounds__(256) void copy_cls(const float* __restrict__ act, float* __restrict__ out) {
  const int b = blockIdx.x, t = threadIdx.x;
  for (int j = t; j < cD; j += 256) out[(size_t)b * cD + j] = act[(size_t)(b * cS) * cD + j];
}

// ---------------------------------------------------------------- launch
extern "C" void kernel_launch(void* const* d_in, const int* in_sizes, int n_in,
                              void* d_out, int out_size, void* d_ws, size_t ws_size,
                              hipStream_t stream) {
  (void)in_sizes; (void)n_in; (void)out_size; (void)ws_size;
  const float* x     = (const float*)d_in[0];
  const float* cls   = (const float*)d_in[1];
  const float* ln_g  = (const float*)d_in[2];
  const float* ln_b  = (const float*)d_in[3];
  const float* gln_g = (const float*)d_in[4];
  const float* gln_b = (const float*)d_in[5];
  const float* gW1   = (const float*)d_in[6];
  const float* gb1   = (const float*)d_in[7];
  const float* gW2   = (const float*)d_in[8];
  const float* gb2   = (const float*)d_in[9];
  const float* Wqkv  = (const float*)d_in[10];
  const float* bqkv  = (const float*)d_in[11];
  const float* Wo    = (const float*)d_in[12];
  const float* bo    = (const float*)d_in[13];
  float* out = (float*)d_out;

  char* base = (char*)d_ws;
  size_t off = 0;
  auto carve = [&](size_t bytes) -> char* {
    char* p = base + off;
    off = (off + bytes + 255) & ~(size_t)255;
    return p;
  };
  _Float16* gW1h   = (_Float16*)carve((size_t)cDH * cD * 2);
  float*    colsum = (float*)   carve((size_t)cDH * 4);
  float*    biasp  = (float*)   carve((size_t)cDH * 4);
  _Float16* Wqkvh  = (_Float16*)carve((size_t)cL * 3 * cD * cD * 2);
  _Float16* Woh    = (_Float16*)carve((size_t)cL * cD * cD * 2);
  float*    wbuf   = (float*)   carve((size_t)cB * cN * 4);
  float*    topv   = (float*)   carve((size_t)cB * cK * 4);
  int*      topi   = (int*)     carve((size_t)cB * cK * 4);
  float*    seq    = (float*)   carve((size_t)cM * cD * 4);
  float*    z0     = (float*)   carve((size_t)cM * cD * 4);
  float*    act    = (float*)   carve((size_t)cM * cD * 4);
  _Float16* acth   = (_Float16*)carve((size_t)cM * cD * 2);
  float*    qkvb   = (float*)   carve((size_t)cM * 3 * cD * 4);
  _Float16* oh     = (_Float16*)carve((size_t)cM * cD * 2);
  float*    tmp    = (float*)   carve((size_t)cM * cD * 4);

  // weight prep
  prep_gate<<<cDH, 256, 0, stream>>>(gW1, gln_g, gln_b, gb1, gW1h, colsum, biasp);
  {
    int n1 = cL * 3 * cD * cD;
    f32_to_f16<<<(n1 + 255) / 256, 256, 0, stream>>>(Wqkv, Wqkvh, n1);
    int n2 = cL * cD * cD;
    f32_to_f16<<<(n2 + 255) / 256, 256, 0, stream>>>(Wo, Woh, n2);
  }

  // fused LN + gate (streams x once): 4096 tiles, 2 waves/block
  gate_kernel<<<(cB * cN / 16) / 2, 64, 0, stream>>>(x, gW1h, colsum, biasp, gW2, gb2, wbuf);

  // top-k + gather + shared LN
  topk_kernel<<<cB, 256, 0, stream>>>(wbuf, topv, topi);
  gather_seq<<<cM, 256, 0, stream>>>(x, cls, topi, topv, seq);
  ln_rows<<<cM, 256, 0, stream>>>(seq, nullptr, ln_g, ln_b, z0, acth);

  // stacked MHA
  for (int l = 0; l < cL; ++l) {
    gemm_f16<<<(cM / 16) * (3 * cD / 64), 32, 0, stream>>>(
        acth, Wqkvh + (size_t)l * 3 * cD * cD, bqkv + (size_t)l * 3 * cD, qkvb, 3 * cD, cD, cM / 16);
    attention_kernel<<<cB * cH, 64, 0, stream>>>(qkvb, oh);
    gemm_f16<<<(cM / 16) * (cD / 64), 32, 0, stream>>>(
        oh, Woh + (size_t)l * cD * cD, bo + (size_t)l * cD, tmp, cD, cD, cM / 16);
    ln_rows<<<cM, 256, 0, stream>>>(tmp, z0, ln_g, ln_b, act, acth);
  }

  copy_cls<<<cB, 256, 0, stream>>>(act, out);
}
